// LocalAttention_29463475651336
// MI455X (gfx1250) — matrix operand order repaired and checked by reading, not compiled, not run on hardware
//
#include <hip/hip_runtime.h>
#include <hip/hip_bf16.h>
#include <stdint.h>

#define D_MODEL  1024
#define N_HEADS  16
#define HEAD_DIM 64
#define HALF_WIN 64
#define B_BATCH  2
#define SEQ_N    2048
#define SEQ_M    2048

typedef __attribute__((ext_vector_type(16))) __bf16       v16bf;
typedef __attribute__((ext_vector_type(8)))  float        v8f;
typedef __attribute__((ext_vector_type(4)))  unsigned int u32x4;
typedef __attribute__((ext_vector_type(2)))  unsigned int u32x2;

union FragAB { u32x4 q[2]; v16bf v; };
union AccV8  { v8f v; float f[8]; };

__device__ __forceinline__ uint32_t pack2_bf16(float a, float b) {
  uint32_t ua = __float_as_uint(a), ub = __float_as_uint(b);
  ua = (ua + 0x7FFFu + ((ua >> 16) & 1u)) >> 16;
  ub = (ub + 0x7FFFu + ((ub >> 16) & 1u)) >> 16;
  return ua | (ub << 16);
}
__device__ __forceinline__ uint16_t f2bf(float a) {
  uint32_t ua = __float_as_uint(a);
  return (uint16_t)((ua + 0x7FFFu + ((ua >> 16) & 1u)) >> 16);
}

// CDNA5 async global->LDS copy (ASYNCcnt-tracked, bypasses VGPRs).
// VDST = per-lane LDS byte address (low 32 bits of the generic shared pointer),
// VADDR = 64-bit global address, GV mode (saddr = off).
__device__ __forceinline__ void async_copy_b128(void* lds_dst, const void* gsrc) {
  asm volatile("global_load_async_to_lds_b128 %0, %1, off"
               :: "v"((uint32_t)(uintptr_t)lds_dst),
                  "v"((uint64_t)(uintptr_t)gsrc)
               : "memory");
}
__device__ __forceinline__ void wait_async_le8() {
  asm volatile("s_wait_asynccnt 0x8" ::: "memory");
}
__device__ __forceinline__ void wait_async_0() {
  asm volatile("s_wait_asynccnt 0x0" ::: "memory");
}

// One-shot fp32 -> bf16 conversion (bandwidth-bound, hoists all conversion
// VALU work out of the GEMM inner loops).  n4 = element count / 4.
__global__ __launch_bounds__(256)
void cvt_f32_bf16(const float* __restrict__ src, uint16_t* __restrict__ dst, int n4) {
  int i = blockIdx.x * 256 + threadIdx.x;
  if (i < n4) {
    float4 x = ((const float4*)src)[i];
    u32x2 o;
    o.x = pack2_bf16(x.x, x.y);
    o.y = pack2_bf16(x.z, x.w);
    *(u32x2*)(dst + (size_t)i * 4) = o;
  }
}

// C[row, col] = A[row, :] . W[col, :] + bias[col]   (A, W bf16; K-dim inner)
// Block tile 128x128, 4 waves (2x2), wave tile 64x64 = 4x4 WMMA 16x16x32.
// Double-buffered LDS staged with global_load_async_to_lds_b128.
template <int OUT_BF16>
__global__ __launch_bounds__(128)
void gemm_bias_wmma(const uint16_t* __restrict__ A, const uint16_t* __restrict__ W,
                    const float* __restrict__ bias, void* __restrict__ Out,
                    int Nrows, int K, int Ocols) {
  __shared__ uint16_t As[2][128 * 32];   // [m][k] bf16
  __shared__ uint16_t Bs[2][128 * 32];   // [n][k] bf16 (n = output col)

  const int tid  = threadIdx.x;
  const int lane = tid & 31;
  const int wid  = tid >> 5;
  const int wm   = (wid >> 1) * 64;
  const int wn   = (wid & 1) * 64;
  const int ml   = lane & 15;
  const int half = lane >> 4;
  const int rowBase = blockIdx.y * 128;
  const int colBase = blockIdx.x * 128;

  // staging map: 512 b128 lines per 128x32 tile -> 4 per thread (x2 tiles = 8 async/thread)
  const int srow = (tid * 4) >> 2;         // == tid; kept explicit below
  (void)srow;

  AccV8 acc[4][4];
  #pragma unroll
  for (int i = 0; i < 4; i++)
    #pragma unroll
    for (int j = 0; j < 4; j++)
      #pragma unroll
      for (int r = 0; r < 8; r++) acc[i][j].f[r] = 0.0f;

  const int nsteps = K / 32;

  // ---- prologue: stage k-step 0 into buffer 0 ----
  #pragma unroll
  for (int t = 0; t < 4; t++) {
    int v = tid + t * 128;                 // 0..511
    int row = v >> 2, c8 = (v & 3) * 8;    // row 0..127, col 0/8/16/24
    async_copy_b128(&As[0][row * 32 + c8], A + (size_t)(rowBase + row) * K + c8);
    async_copy_b128(&Bs[0][row * 32 + c8], W + (size_t)(colBase + row) * K + c8);
  }

  for (int s = 0; s < nsteps; s++) {
    const int cur = s & 1;
    if (s + 1 < nsteps) {
      const int k1 = (s + 1) * 32;
      #pragma unroll
      for (int t = 0; t < 4; t++) {
        int v = tid + t * 128;
        int row = v >> 2, c8 = (v & 3) * 8;
        async_copy_b128(&As[cur ^ 1][row * 32 + c8],
                        A + (size_t)(rowBase + row) * K + k1 + c8);
        async_copy_b128(&Bs[cur ^ 1][row * 32 + c8],
                        W + (size_t)(colBase + row) * K + k1 + c8);
      }
      wait_async_le8();   // in-order completion: stage s (oldest 8) now done
    } else {
      wait_async_0();
    }
    __syncthreads();      // stage s visible to all waves

    FragAB af[4], bfr[4];
    #pragma unroll
    for (int i = 0; i < 4; i++) {
      const uint16_t* pr = &As[cur][(wm + i * 16 + ml) * 32];
      af[i].q[0] = *(const u32x4*)(pr + half * 8);        // K = half*8 .. +7
      af[i].q[1] = *(const u32x4*)(pr + 16 + half * 8);   // K = 16+half*8 .. +7
    }
    #pragma unroll
    for (int j = 0; j < 4; j++) {
      const uint16_t* pr = &Bs[cur][(wn + j * 16 + ml) * 32];
      bfr[j].q[0] = *(const u32x4*)(pr + half * 8);       // K = half*8 .. +7
      bfr[j].q[1] = *(const u32x4*)(pr + 16 + half * 8);  // K = 16+half*8 .. +7
    }
    #pragma unroll
    for (int i = 0; i < 4; i++)
      #pragma unroll
      for (int j = 0; j < 4; j++)
        acc[i][j].v = __builtin_amdgcn_wmma_f32_16x16x32_bf16(
            false, af[i].v, false, bfr[j].v, (short)0, acc[i][j].v, false, false);
    __syncthreads();      // all waves done reading buffer cur before it is refilled
  }

  // ---- epilogue: bias add + store (C layout: lane=N, VGPR r -> M=r+half*8) ----
  #pragma unroll
  for (int j = 0; j < 4; j++) {
    int col  = colBase + wn + j * 16 + ml;
    float bv = bias[col];
    #pragma unroll
    for (int i = 0; i < 4; i++) {
      #pragma unroll
      for (int r = 0; r < 8; r++) {
        int row   = rowBase + wm + i * 16 + half * 8 + r;
        float val = acc[i][j].f[r] + bv;
        if (OUT_BF16) ((uint16_t*)Out)[(size_t)row * Ocols + col] = f2bf(val);
        else          ((float*)Out)[(size_t)row * Ocols + col]    = val;
      }
    }
  }
}

// Windowed attention: one block per (batch, head, 64-query tile); 4 waves x 16 rows.
// Key span = [qt-64, qt+127] (192 keys), S/P via WMMA, row softmax via shfl.
__global__ __launch_bounds__(128)
void local_attn_wmma(const uint16_t* __restrict__ Qb, const uint16_t* __restrict__ Kb,
                     const uint16_t* __restrict__ Vb, uint16_t* __restrict__ Ob) {
  __shared__ uint16_t sK[192 * 64];   // [key][d]; reused later as P [64 rows][192 keys]
  __shared__ uint16_t sVT[64 * 192];  // [d][key]

  const int tid  = threadIdx.x;
  const int lane = tid & 31;
  const int wid  = tid >> 5;
  const int ml   = lane & 15;
  const int half = lane >> 4;

  const int qt    = blockIdx.x * 64;
  const int h     = blockIdx.y;
  const int bz    = blockIdx.z;
  const int kbase = qt - HALF_WIN;

  const size_t qoff = (size_t)bz * SEQ_N * D_MODEL + h * HEAD_DIM;
  const size_t koff = (size_t)bz * SEQ_M * D_MODEL + h * HEAD_DIM;

  // ---- stage K [192][64] (zero-fill out-of-range keys) ----
  for (int e = tid; e < 192 * 16; e += 128) {
    int key = e >> 4, d4 = e & 15;
    int kg  = kbase + key;
    u32x2 val; val.x = 0u; val.y = 0u;
    if (kg >= 0 && kg < SEQ_M)
      val = *(const u32x2*)(Kb + koff + (size_t)kg * D_MODEL + d4 * 4);
    *(u32x2*)(sK + key * 64 + d4 * 4) = val;
  }
  // ---- stage V^T [64][192] ----
  for (int e = tid; e < 192 * 64; e += 128) {
    int key = e >> 6, d = e & 63;
    int kg  = kbase + key;
    uint16_t val = 0;
    if (kg >= 0 && kg < SEQ_M)
      val = Vb[koff + (size_t)kg * D_MODEL + d];
    sVT[d * 192 + key] = val;
  }
  __syncthreads();

  // ---- Q fragments for this wave's 16 rows (hd=64 -> two K-steps of 32) ----
  FragAB aq[2];
  {
    const uint16_t* p = Qb + qoff + (size_t)(qt + wid * 16 + ml) * D_MODEL;
    #pragma unroll
    for (int kk = 0; kk < 2; kk++) {
      aq[kk].q[0] = *(const u32x4*)(p + kk * 32 + half * 8);
      aq[kk].q[1] = *(const u32x4*)(p + kk * 32 + 16 + half * 8);
    }
  }

  // ---- S = Q K^T over 12 key tiles ----
  AccV8 s[12];
  #pragma unroll
  for (int j = 0; j < 12; j++) {
    #pragma unroll
    for (int r = 0; r < 8; r++) s[j].f[r] = 0.0f;
    #pragma unroll
    for (int kk = 0; kk < 2; kk++) {
      FragAB bk;
      const uint16_t* pr = sK + (j * 16 + ml) * 64 + kk * 32;
      bk.q[0] = *(const u32x4*)(pr + half * 8);
      bk.q[1] = *(const u32x4*)(pr + 16 + half * 8);
      s[j].v = __builtin_amdgcn_wmma_f32_16x16x32_bf16(
          false, aq[kk].v, false, bk.v, (short)0, s[j].v, false, false);
    }
  }

  // ---- mask + scale in C-layout registers ----
  const float scale = 0.125f;  // 64^-0.5
  const int qrow = qt + wid * 16 + half * 8;
  #pragma unroll
  for (int j = 0; j < 12; j++) {
    int key = kbase + j * 16 + ml;
    bool kvalid = (key >= 0) && (key < SEQ_M);
    #pragma unroll
    for (int r = 0; r < 8; r++) {
      int dlt = (qrow + r) - key;
      bool ok = kvalid && (dlt <= HALF_WIN) && (dlt >= -HALF_WIN);
      s[j].f[r] = ok ? s[j].f[r] * scale : -3.0e38f;
    }
  }

  // ---- row softmax: reduce over tiles then across 16 lanes of each half ----
  float mx[8], sm[8];
  #pragma unroll
  for (int r = 0; r < 8; r++) {
    float m = s[0].f[r];
    #pragma unroll
    for (int j = 1; j < 12; j++) m = fmaxf(m, s[j].f[r]);
    mx[r] = m;
  }
  #pragma unroll
  for (int off = 8; off >= 1; off >>= 1)
    #pragma unroll
    for (int r = 0; r < 8; r++) mx[r] = fmaxf(mx[r], __shfl_xor(mx[r], off, 16));
  #pragma unroll
  for (int r = 0; r < 8; r++) sm[r] = 0.0f;
  #pragma unroll
  for (int j = 0; j < 12; j++)
    #pragma unroll
    for (int r = 0; r < 8; r++) {
      float e = __expf(s[j].f[r] - mx[r]);
      s[j].f[r] = e;
      sm[r] += e;
    }
  #pragma unroll
  for (int off = 8; off >= 1; off >>= 1)
    #pragma unroll
    for (int r = 0; r < 8; r++) sm[r] += __shfl_xor(sm[r], off, 16);

  __syncthreads();  // all waves finished reading sK; reuse it for P
  #pragma unroll
  for (int j = 0; j < 12; j++)
    #pragma unroll
    for (int r = 0; r < 8; r++)
      sK[(wid * 16 + half * 8 + r) * 192 + j * 16 + ml] = f2bf(s[j].f[r]);
  __syncthreads();

  // ---- O = P V  (K-dim 192 -> 6 WMMA K-steps; 4 output d-tiles) ----
  AccV8 o[4];
  #pragma unroll
  for (int dj = 0; dj < 4; dj++)
    #pragma unroll
    for (int r = 0; r < 8; r++) o[dj].f[r] = 0.0f;
  #pragma unroll
  for (int ks = 0; ks < 6; ks++) {
    FragAB ap;
    const uint16_t* pr = sK + (wid * 16 + ml) * 192 + ks * 32;
    ap.q[0] = *(const u32x4*)(pr + half * 8);
    ap.q[1] = *(const u32x4*)(pr + 16 + half * 8);
    #pragma unroll
    for (int dj = 0; dj < 4; dj++) {
      FragAB bv;
      const uint16_t* pv = sVT + (dj * 16 + ml) * 192 + ks * 32;
      bv.q[0] = *(const u32x4*)(pv + half * 8);
      bv.q[1] = *(const u32x4*)(pv + 16 + half * 8);
      o[dj].v = __builtin_amdgcn_wmma_f32_16x16x32_bf16(
          false, ap.v, false, bv.v, (short)0, o[dj].v, false, false);
    }
  }

  // ---- normalize by row sums and store bf16 ----
  #pragma unroll
  for (int dj = 0; dj < 4; dj++)
    #pragma unroll
    for (int r = 0; r < 8; r++) {
      int q = qt + wid * 16 + half * 8 + r;
      int d = dj * 16 + ml;
      Ob[qoff + (size_t)q * D_MODEL + d] = f2bf(o[dj].f[r] / sm[r]);
    }
}

extern "C" void kernel_launch(void* const* d_in, const int* in_sizes, int n_in,
                              void* d_out, int out_size, void* d_ws, size_t ws_size,
                              hipStream_t stream) {
  (void)in_sizes; (void)n_in; (void)out_size; (void)ws_size;
  const float* queries = (const float*)d_in[0];
  const float* keys    = (const float*)d_in[1];
  const float* values  = (const float*)d_in[2];
  const float* wq = (const float*)d_in[3];
  const float* bq = (const float*)d_in[4];
  const float* wk = (const float*)d_in[5];
  const float* bk = (const float*)d_in[6];
  const float* wv = (const float*)d_in[7];
  const float* bv = (const float*)d_in[8];
  const float* wo = (const float*)d_in[9];
  const float* bo = (const float*)d_in[10];

  const int Nrows = B_BATCH * SEQ_N;               // 4096
  const size_t actE = (size_t)Nrows * D_MODEL;     // 4,194,304
  const size_t wE   = (size_t)D_MODEL * D_MODEL;   // 1,048,576

  uint16_t* Qb  = (uint16_t*)d_ws;                 // bf16 projections
  uint16_t* Kb  = Qb + actE;
  uint16_t* Vb  = Kb + actE;
  uint16_t* Ob  = Vb + actE;
  uint16_t* Xq  = Ob + actE;                       // bf16 activations
  uint16_t* Xk  = Xq + actE;
  uint16_t* Xv  = Xk + actE;
  uint16_t* Wqb = Xv + actE;                       // bf16 weights
  uint16_t* Wkb = Wqb + wE;
  uint16_t* Wvb = Wkb + wE;
  uint16_t* Wob = Wvb + wE;                        // ~67 MB total

  // one-shot conversions (bandwidth-bound)
  const int a4 = (int)(actE / 4), w4 = (int)(wE / 4);
  dim3 ca((a4 + 255) / 256), cw((w4 + 255) / 256);
  cvt_f32_bf16<<<ca, 256, 0, stream>>>(queries, Xq, a4);
  cvt_f32_bf16<<<ca, 256, 0, stream>>>(keys,    Xk, a4);
  cvt_f32_bf16<<<ca, 256, 0, stream>>>(values,  Xv, a4);
  cvt_f32_bf16<<<cw, 256, 0, stream>>>(wq, Wqb, w4);
  cvt_f32_bf16<<<cw, 256, 0, stream>>>(wk, Wkb, w4);
  cvt_f32_bf16<<<cw, 256, 0, stream>>>(wv, Wvb, w4);
  cvt_f32_bf16<<<cw, 256, 0, stream>>>(wo, Wob, w4);

  dim3 gg(D_MODEL / 128, Nrows / 128);
  gemm_bias_wmma<1><<<gg, 128, 0, stream>>>(Xq, Wqb, bq, Qb, Nrows, D_MODEL, D_MODEL);
  gemm_bias_wmma<1><<<gg, 128, 0, stream>>>(Xk, Wkb, bk, Kb, Nrows, D_MODEL, D_MODEL);
  gemm_bias_wmma<1><<<gg, 128, 0, stream>>>(Xv, Wvb, bv, Vb, Nrows, D_MODEL, D_MODEL);

  dim3 ga(SEQ_N / 64, N_HEADS, B_BATCH);
  local_attn_wmma<<<ga, 128, 0, stream>>>(Qb, Kb, Vb, Ob);

  gemm_bias_wmma<0><<<gg, 128, 0, stream>>>(Ob, Wob, bo, d_out, Nrows, D_MODEL, D_MODEL);
}
